// BlockNTPTransformer_43319040147986
// MI455X (gfx1250) — matrix operand order repaired and checked by reading, not compiled, not run on hardware
//
#include <hip/hip_runtime.h>
#include <hip/hip_bf16.h>
#include <math.h>

// ---------------- problem constants (match reference) ----------------
#define BT   2
#define TT   512
#define SS   1024          // 2*T
#define DD   1024
#define HH   16
#define DHH  64
#define DFFN 4096
#define VV   32000
#define CSLL 16
#define NLB  4
#define NLD  2
#define VCH  3200          // vocab chunk (multiple of 64), 10 chunks

typedef __attribute__((ext_vector_type(16))) _Float16 v16h;
typedef __attribute__((ext_vector_type(8)))  _Float16 h8;
typedef __attribute__((ext_vector_type(4)))  _Float16 h4;
typedef __attribute__((ext_vector_type(2)))  _Float16 h2;
typedef __attribute__((ext_vector_type(8)))  float    v8f;
typedef __attribute__((ext_vector_type(4)))  float    f4;

// =====================================================================
// WMMA GEMM:  C[M,N] = epi( alpha * A[M,K] @ op(B) )
//   TRANSB==0: B is [K,N] (ldb); TRANSB==1: B is [N,K] (ldb), compute A@B^T
//   EPI: 0 = none, 1 = +bias, 2 = +bias+residual, 3 = gelu(.+bias)
// Block: 128 threads = 4 waves. Block tile 64x64, K-step 32.
// Wave tile 32x32 -> 4 accumulators, 4 v_wmma_f32_16x16x32_f16 / k-step,
// 8 ds_load_b128 fragment reads / k-step (2 per WMMA).
// LDS rows stride 40 f16 (80B): 16B-aligned so fragments load as b128;
// 20-dword row stride => conflict-free b128 per 16-lane half.
// Requires M%64==0, N%64==0, K%32==0, 16B-aligned pointers, lda/ldb%4==0.
// =====================================================================
#define LDSK 40

template<int TRANSB, int EPI>
__global__ __launch_bounds__(128)
void gemm_wmma(const float* __restrict__ A, int lda,
               const float* __restrict__ B, int ldb,
               const float* __restrict__ bias,
               const float* __restrict__ residual,
               float* __restrict__ C, int ldc,
               int M, int N, int K, float alpha)
{
    __shared__ alignas(16) _Float16 As[64 * LDSK];
    __shared__ alignas(16) _Float16 Bs[64 * LDSK];

    const int tid  = threadIdx.x;
    const int lane = tid & 31;
    const int wave = tid >> 5;        // 0..3
    const int wr   = wave >> 1;       // 0..1 : 32-row tile
    const int wc   = wave & 1;        // 0..1 : 32-col tile
    const int hsel = lane >> 4;       // half-wave select
    const int rc   = lane & 15;       // row (A frag) / col (B frag)
    const int n0   = blockIdx.x * 64;
    const int m0   = blockIdx.y * 64;

    v8f acc[2][2] = {};

    for (int k0 = 0; k0 < K; k0 += 32) {
        // ---- stage A: 64x32 fp32 -> f16, float4 granularity (512 quads) ----
#pragma unroll
        for (int it = 0; it < 4; ++it) {
            int q  = tid + it * 128;          // 0..511
            int m  = q >> 3;                  // 0..63
            int kq = q & 7;                   // quad-of-4 within K-step
            f4 v = *(const f4*)(A + (size_t)(m0 + m) * lda + k0 + kq * 4);
            h4 hv = { (_Float16)v[0], (_Float16)v[1], (_Float16)v[2], (_Float16)v[3] };
            *(h4*)(&As[m * LDSK + kq * 4]) = hv;   // ds_store_b64
        }
        // ---- stage B as [n][k] ----
        if (TRANSB) {
#pragma unroll
            for (int it = 0; it < 4; ++it) {
                int q  = tid + it * 128;
                int n  = q >> 3;
                int kq = q & 7;
                f4 v = *(const f4*)(B + (size_t)(n0 + n) * ldb + k0 + kq * 4);
                h4 hv = { (_Float16)v[0], (_Float16)v[1], (_Float16)v[2], (_Float16)v[3] };
                *(h4*)(&Bs[n * LDSK + kq * 4]) = hv;
            }
        } else {
            // B row-major [K,N]: load two K-rows x 4 N, store k-pairs as b32
            // into the transposed LDS tile. 256 tasks.
#pragma unroll
            for (int it = 0; it < 2; ++it) {
                int q  = tid + it * 128;      // 0..255
                int k2 = q >> 4;              // 0..15 (pair of K rows)
                int nq = q & 15;              // 0..15 (quad of N)
                const float* bp = B + (size_t)(k0 + 2 * k2) * ldb + n0 + nq * 4;
                f4 v0 = *(const f4*)bp;
                f4 v1 = *(const f4*)(bp + ldb);
#pragma unroll
                for (int j = 0; j < 4; ++j) {
                    h2 p = { (_Float16)v0[j], (_Float16)v1[j] };
                    *(h2*)(&Bs[(nq * 4 + j) * LDSK + 2 * k2]) = p;  // ds_store_b32
                }
            }
        }
        __syncthreads();

        // ---- fragment loads (CDNA5 16-bit layouts, ISA 7.12.2) ----
        // A 16x32: lane half h, row rc: f16[0..7]=K[8h..8h+7], f16[8..15]=K[8h+16..8h+23]
        // B 32x16: lane half h, col rc: f16[0..7]=K[8h..8h+7], f16[8..15]=K[8h+16..8h+23]
        const _Float16* pa0 = &As[(wr * 32 + rc) * LDSK + 8 * hsel];
        const _Float16* pa1 = &As[(wr * 32 + 16 + rc) * LDSK + 8 * hsel];
        const _Float16* pb0 = &Bs[(wc * 32 + rc) * LDSK + 8 * hsel];
        const _Float16* pb1 = &Bs[(wc * 32 + 16 + rc) * LDSK + 8 * hsel];
        h8 a0l = *(const h8*)(pa0),     a0h = *(const h8*)(pa0 + 16);
        h8 a1l = *(const h8*)(pa1),     a1h = *(const h8*)(pa1 + 16);
        h8 b0l = *(const h8*)(pb0),     b0h = *(const h8*)(pb0 + 16);
        h8 b1l = *(const h8*)(pb1),     b1h = *(const h8*)(pb1 + 16);
        v16h a0 = __builtin_shufflevector(a0l, a0h, 0,1,2,3,4,5,6,7,8,9,10,11,12,13,14,15);
        v16h a1 = __builtin_shufflevector(a1l, a1h, 0,1,2,3,4,5,6,7,8,9,10,11,12,13,14,15);
        v16h b0 = __builtin_shufflevector(b0l, b0h, 0,1,2,3,4,5,6,7,8,9,10,11,12,13,14,15);
        v16h b1 = __builtin_shufflevector(b1l, b1h, 0,1,2,3,4,5,6,7,8,9,10,11,12,13,14,15);

        acc[0][0] = __builtin_amdgcn_wmma_f32_16x16x32_f16(false, a0, false, b0, (short)0, acc[0][0], false, false);
        acc[0][1] = __builtin_amdgcn_wmma_f32_16x16x32_f16(false, a0, false, b1, (short)0, acc[0][1], false, false);
        acc[1][0] = __builtin_amdgcn_wmma_f32_16x16x32_f16(false, a1, false, b0, (short)0, acc[1][0], false, false);
        acc[1][1] = __builtin_amdgcn_wmma_f32_16x16x32_f16(false, a1, false, b1, (short)0, acc[1][1], false, false);
        __syncthreads();
    }

    // ---- epilogue: C/D layout -> VGPR e holds (M = e + 8*hsel, N = rc) ----
#pragma unroll
    for (int j = 0; j < 2; ++j) {
        const int col = n0 + wc * 32 + j * 16 + rc;
        const float bv = (EPI >= 1) ? bias[col] : 0.0f;   // hoisted, 1 load/tile-col
#pragma unroll
        for (int i = 0; i < 2; ++i) {
            const int rbase = m0 + wr * 32 + i * 16 + hsel * 8;
            float v[8];
#pragma unroll
            for (int e = 0; e < 8; ++e)
                v[e] = acc[i][j][e] * alpha + bv;
            if (EPI == 2) {
                float r[8];
#pragma unroll
                for (int e = 0; e < 8; ++e)           // batched loads
                    r[e] = residual[(size_t)(rbase + e) * ldc + col];
#pragma unroll
                for (int e = 0; e < 8; ++e)
                    v[e] += r[e];
            }
            if (EPI == 3) {
#pragma unroll
                for (int e = 0; e < 8; ++e)
                    v[e] = 0.5f * v[e] * (1.0f + erff(v[e] * 0.70710678118f));
            }
#pragma unroll
            for (int e = 0; e < 8; ++e)
                C[(size_t)(rbase + e) * ldc + col] = v[e];
        }
    }
}

// =====================================================================
// LayerNorm: one block per row of length Dn
// =====================================================================
__global__ __launch_bounds__(256)
void layernorm_k(const float* __restrict__ x, const float* __restrict__ g,
                 const float* __restrict__ b, float* __restrict__ out, int Dn)
{
    __shared__ float red[256];
    const int row = blockIdx.x;
    const float* xr = x + (size_t)row * Dn;

    float s = 0.f;
    for (int i = threadIdx.x; i < Dn; i += 256) s += xr[i];
    red[threadIdx.x] = s; __syncthreads();
    for (int st = 128; st > 0; st >>= 1) {
        if (threadIdx.x < st) red[threadIdx.x] += red[threadIdx.x + st];
        __syncthreads();
    }
    float mu = red[0] / Dn;
    __syncthreads();

    float v = 0.f;
    for (int i = threadIdx.x; i < Dn; i += 256) { float d = xr[i] - mu; v += d * d; }
    red[threadIdx.x] = v; __syncthreads();
    for (int st = 128; st > 0; st >>= 1) {
        if (threadIdx.x < st) red[threadIdx.x] += red[threadIdx.x + st];
        __syncthreads();
    }
    float inv = rsqrtf(red[0] / Dn + 1e-5f);

    float* outr = out + (size_t)row * Dn;
    for (int i = threadIdx.x; i < Dn; i += 256)
        outr[i] = (xr[i] - mu) * inv * g[i] + b[i];
}

// =====================================================================
// Fused mask + softmax. Masks in the reference are row-constant:
//  maskType 0 (block_ntp): row fully open iff  r<480 || r>=512
//  maskType 1 (ar):        row fully open iff  r<=510 || (r>=512 && (r-512)%16!=15)
// Fully-masked row (all -1e9) softmaxes to uniform 1/S.
// =====================================================================
__global__ __launch_bounds__(256)
void softmax_mask_k(float* __restrict__ sc, int S, int maskType)
{
    const int r = blockIdx.x;
    bool active;
    if (maskType == 0) active = (r < 480) || (r >= 512);
    else               active = (r <= 510) || (r >= 512 && ((r - 512) & 15) != 15);

    float* row = sc + (size_t)r * S;
    __shared__ float red[256];

    if (!active) {
        float u = 1.0f / (float)S;
        for (int i = threadIdx.x; i < S; i += 256) row[i] = u;
        return;
    }
    float m = -3.4e38f;
    for (int i = threadIdx.x; i < S; i += 256) m = fmaxf(m, row[i]);
    red[threadIdx.x] = m; __syncthreads();
    for (int st = 128; st > 0; st >>= 1) {
        if (threadIdx.x < st) red[threadIdx.x] = fmaxf(red[threadIdx.x], red[threadIdx.x + st]);
        __syncthreads();
    }
    m = red[0]; __syncthreads();

    float s = 0.f;
    for (int i = threadIdx.x; i < S; i += 256) {
        float e = expf(row[i] - m); row[i] = e; s += e;
    }
    red[threadIdx.x] = s; __syncthreads();
    for (int st = 128; st > 0; st >>= 1) {
        if (threadIdx.x < st) red[threadIdx.x] += red[threadIdx.x + st];
        __syncthreads();
    }
    float inv = 1.0f / red[0];
    for (int i = threadIdx.x; i < S; i += 256) row[i] *= inv;
}

// =====================================================================
// Embedding: x[b,s,:] = (s<T ? tok_emb[ids[b,s]] : mask_tokens[(s-T)%CSL])
//                       + pos_emb[s % T]
// =====================================================================
__global__ void embed_k(const int* __restrict__ ids,
                        const float* __restrict__ tok_emb,
                        const float* __restrict__ pos_emb,
                        const float* __restrict__ mask_tokens,
                        float* __restrict__ x)
{
    int idx = blockIdx.x * 256 + threadIdx.x;     // over BT*SS*DD
    if (idx >= BT * SS * DD) return;
    int d = idx % DD;
    int s = (idx / DD) % SS;
    int b = idx / (DD * SS);
    float e;
    if (s < TT)
        e = tok_emb[(size_t)ids[b * TT + s] * DD + d] + pos_emb[(size_t)s * DD + d];
    else
        e = mask_tokens[(size_t)((s - TT) % CSLL) * DD + d] + pos_emb[(size_t)(s - TT) * DD + d];
    x[idx] = e;
}

// Gather the 1022 rows that feed the loss into a padded [1024, D] matrix.
__global__ void gather_rows_k(const float* __restrict__ x, float* __restrict__ Ac)
{
    int idx = blockIdx.x * 256 + threadIdx.x;     // 1024*DD
    if (idx >= 1024 * DD) return;
    int d = idx % DD;
    int r = idx / DD;
    float v = 0.f;
    if (r < 1022) {
        int b = r / 511, j = r % 511;
        v = x[(size_t)(b * SS + TT + 1 + j) * DD + d];
    }
    Ac[idx] = v;
}

__global__ void lse_init_k(float* rmax, float* rsum, float* rtgt)
{
    int i = blockIdx.x * 256 + threadIdx.x;
    if (i < 1024) { rmax[i] = -3.4e38f; rsum[i] = 0.f; rtgt[i] = 0.f; }
}

// Online logsumexp over one vocab chunk + capture target logit.
__global__ __launch_bounds__(256)
void lse_update_k(const float* __restrict__ Cchunk, int chunkBase, int chunkW,
                  const int* __restrict__ ids,
                  float* __restrict__ rmax, float* __restrict__ rsum,
                  float* __restrict__ rtgt)
{
    const int r = blockIdx.x;                     // 0..1021
    const float* row = Cchunk + (size_t)r * chunkW;
    __shared__ float red[256];

    float m = -3.4e38f;
    for (int i = threadIdx.x; i < chunkW; i += 256) m = fmaxf(m, row[i]);
    red[threadIdx.x] = m; __syncthreads();
    for (int st = 128; st > 0; st >>= 1) {
        if (threadIdx.x < st) red[threadIdx.x] = fmaxf(red[threadIdx.x], red[threadIdx.x + st]);
        __syncthreads();
    }
    m = red[0]; __syncthreads();

    float s = 0.f;
    for (int i = threadIdx.x; i < chunkW; i += 256) s += expf(row[i] - m);
    red[threadIdx.x] = s; __syncthreads();
    for (int st = 128; st > 0; st >>= 1) {
        if (threadIdx.x < st) red[threadIdx.x] += red[threadIdx.x + st];
        __syncthreads();
    }
    s = red[0];

    if (threadIdx.x == 0) {
        float om = rmax[r], os = rsum[r];
        float nm = fmaxf(om, m);
        rsum[r] = os * expf(om - nm) + s * expf(m - nm);
        rmax[r] = nm;
        int b = r / 511, j = r % 511;
        int tgt = ids[b * TT + j];
        if (tgt >= chunkBase && tgt < chunkBase + chunkW)
            rtgt[r] = row[tgt - chunkBase];
    }
}

__global__ __launch_bounds__(256)
void loss_k(const float* __restrict__ rmax, const float* __restrict__ rsum,
            const float* __restrict__ rtgt, float* __restrict__ out)
{
    __shared__ float red[256];
    float s = 0.f;
    for (int r = threadIdx.x; r < 1022; r += 256)
        s += (rmax[r] + logf(rsum[r])) - rtgt[r];
    red[threadIdx.x] = s; __syncthreads();
    for (int st = 128; st > 0; st >>= 1) {
        if (threadIdx.x < st) red[threadIdx.x] += red[threadIdx.x + st];
        __syncthreads();
    }
    if (threadIdx.x == 0) out[0] = red[0] / 1022.0f;
}

// =====================================================================
// Host orchestration (graph-capture safe: only kernel launches on stream)
// =====================================================================
extern "C" void kernel_launch(void* const* d_in, const int* in_sizes, int n_in,
                              void* d_out, int out_size, void* d_ws, size_t ws_size,
                              hipStream_t stream)
{
    (void)in_sizes; (void)n_in; (void)out_size; (void)ws_size;

    const int*   tok_ids     = (const int*)d_in[0];
    const float* tok_emb     = (const float*)d_in[1];
    const float* pos_emb     = (const float*)d_in[2];
    const float* mask_tokens = (const float*)d_in[3];

    // ---- workspace carve-up ----
    char* wsp = (char*)d_ws;
    size_t off = 0;
    auto alloc = [&](size_t bytes) -> void* {
        void* p = wsp + off;
        off = (off + bytes + 255) & ~(size_t)255;
        return p;
    };
    float* x    = (float*)alloc((size_t)BT * SS * DD * 4);       // residual stream
    float* hbuf = (float*)alloc((size_t)BT * SS * DD * 4);       // LN output
    float* qkv  = (float*)alloc((size_t)BT * SS * 3 * DD * 4);
    float* obuf = (float*)alloc((size_t)BT * SS * DD * 4);       // attn output
    float* ff   = (float*)alloc((size_t)BT * SS * DFFN * 4);     // FFN hidden
    float* sc   = (float*)alloc((size_t)SS * SS * 4);            // per-(b,h) scores
    float* Ac   = (float*)alloc((size_t)1024 * DD * 4);          // loss rows (padded)
    float* lc   = (float*)alloc((size_t)1024 * VCH * 4);         // logits chunk
    float* rmax = (float*)alloc(1024 * 4);
    float* rsum = (float*)alloc(1024 * 4);
    float* rtgt = (float*)alloc(1024 * 4);

    const int MR = BT * SS;   // 2048 rows

    // ---- embeddings ----
    embed_k<<<(BT * SS * DD) / 256, 256, 0, stream>>>(tok_ids, tok_emb, pos_emb,
                                                      mask_tokens, x);

    // ---- 4 body layers (block_ntp mask) + 2 decoder layers (ar mask) ----
    for (int L = 0; L < NLB + NLD; ++L) {
        const int base = (L < NLB) ? 4 : 16;
        const int li   = (L < NLB) ? L : (L - NLB);
        const int maskType = (L < NLB) ? 0 : 1;

        const float* ln1g = (const float*)d_in[base + 0]  + (size_t)li * DD;
        const float* ln1b = (const float*)d_in[base + 1]  + (size_t)li * DD;
        const float* wqkv = (const float*)d_in[base + 2]  + (size_t)li * DD * 3 * DD;
        const float* bqkv = (const float*)d_in[base + 3]  + (size_t)li * 3 * DD;
        const float* wo   = (const float*)d_in[base + 4]  + (size_t)li * DD * DD;
        const float* bo   = (const float*)d_in[base + 5]  + (size_t)li * DD;
        const float* ln2g = (const float*)d_in[base + 6]  + (size_t)li * DD;
        const float* ln2b = (const float*)d_in[base + 7]  + (size_t)li * DD;
        const float* w1   = (const float*)d_in[base + 8]  + (size_t)li * DD * DFFN;
        const float* b1   = (const float*)d_in[base + 9]  + (size_t)li * DFFN;
        const float* w2   = (const float*)d_in[base + 10] + (size_t)li * DFFN * DD;
        const float* b2   = (const float*)d_in[base + 11] + (size_t)li * DD;

        // h = LN1(x);  qkv = h @ wqkv + bqkv
        layernorm_k<<<MR, 256, 0, stream>>>(x, ln1g, ln1b, hbuf, DD);
        gemm_wmma<0,1><<<dim3(3 * DD / 64, MR / 64), 128, 0, stream>>>(
            hbuf, DD, wqkv, 3 * DD, bqkv, nullptr, qkv, 3 * DD, MR, 3 * DD, DD, 1.0f);

        // attention per (batch, head); stream-serialized so `sc` is reused
        for (int b = 0; b < BT; ++b) {
            for (int h = 0; h < HH; ++h) {
                const float* qp = qkv + (size_t)b * SS * 3 * DD + h * DHH;
                const float* kp = qp + DD;
                const float* vp = qp + 2 * DD;
                // scores = (q @ k^T) / sqrt(DH)
                gemm_wmma<1,0><<<dim3(SS / 64, SS / 64), 128, 0, stream>>>(
                    qp, 3 * DD, kp, 3 * DD, nullptr, nullptr, sc, SS, SS, SS, DHH, 0.125f);
                softmax_mask_k<<<SS, 256, 0, stream>>>(sc, SS, maskType);
                // o_bh = attn @ v  (written into [B,S,D] at head column offset)
                float* op = obuf + (size_t)b * SS * DD + h * DHH;
                gemm_wmma<0,0><<<dim3(DHH / 64, SS / 64), 128, 0, stream>>>(
                    sc, SS, vp, 3 * DD, nullptr, nullptr, op, DD, SS, DHH, SS, 1.0f);
            }
        }

        // x = x + o @ wo + bo   (in-place residual: 1 read/write per element)
        gemm_wmma<0,2><<<dim3(DD / 64, MR / 64), 128, 0, stream>>>(
            obuf, DD, wo, DD, bo, x, x, DD, MR, DD, DD, 1.0f);

        // FFN: x = x + gelu(LN2(x) @ w1 + b1) @ w2 + b2
        layernorm_k<<<MR, 256, 0, stream>>>(x, ln2g, ln2b, hbuf, DD);
        gemm_wmma<0,3><<<dim3(DFFN / 64, MR / 64), 128, 0, stream>>>(
            hbuf, DD, w1, DFFN, b1, nullptr, ff, DFFN, MR, DFFN, DD, 1.0f);
        gemm_wmma<0,2><<<dim3(DD / 64, MR / 64), 128, 0, stream>>>(
            ff, DFFN, w2, DD, b2, x, x, DD, MR, DD, DFFN, 1.0f);
    }

    // ---- LM loss without materializing [1022, 32000] logits ----
    gather_rows_k<<<(1024 * DD) / 256, 256, 0, stream>>>(x, Ac);
    lse_init_k<<<4, 256, 0, stream>>>(rmax, rsum, rtgt);
    for (int c = 0; c < VV / VCH; ++c) {          // 10 chunks of 3200
        gemm_wmma<1,0><<<dim3(VCH / 64, 1024 / 64), 128, 0, stream>>>(
            Ac, DD, tok_emb + (size_t)c * VCH * DD, DD, nullptr, nullptr,
            lc, VCH, 1024, VCH, DD, 1.0f);
        lse_update_k<<<1022, 256, 0, stream>>>(lc, c * VCH, VCH, tok_ids,
                                               rmax, rsum, rtgt);
    }
    loss_k<<<1, 256, 0, stream>>>(rmax, rsum, rtgt, (float*)d_out);
}
